// VQFramePrimer_1279900254355
// MI455X (gfx1250) — matrix operand order repaired and checked by reading, not compiled, not run on hardware
//
#include <hip/hip_runtime.h>
#include <hip/hip_bf16.h>

// ---------------- model constants ----------------
#define Bn    2
#define Fn    1024        // BINS
#define Tn    512         // CROP (= sequence length)
#define HEADS 4
#define DHn   256         // head dim
#define FFDn  2048
#define KCn   1024        // codebook size
#define EPSf  1e-5f
#define FT    ((size_t)Fn * (size_t)Tn)

typedef __attribute__((ext_vector_type(16))) __bf16 v16bf;
typedef __attribute__((ext_vector_type(8)))  float  v8f;
typedef int v4i_vs __attribute__((vector_size(16)));   // matches builtin param type

#define AS1 __attribute__((address_space(1)))
#define AS3 __attribute__((address_space(3)))

// CDNA5 async global->LDS staging (compile-safe probe)
#if __has_builtin(__builtin_amdgcn_global_load_async_to_lds_b128)
#define GEMM_ASYNC 1
#else
#define GEMM_ASYNC 0
#endif

#if __has_builtin(__builtin_amdgcn_s_wait_asynccnt)
#define WAIT_ASYNC() __builtin_amdgcn_s_wait_asynccnt(0)
#else
#define WAIT_ASYNC() asm volatile("s_wait_asynccnt 0x0" ::: "memory")
#endif

// ---------------- helpers ----------------
__device__ __forceinline__ unsigned short f2bfu(float f) {
  union { float f; unsigned u; } c; c.f = f;
  unsigned u = c.u;
  unsigned r = u + 0x7FFFu + ((u >> 16) & 1u);   // round-to-nearest-even
  return (unsigned short)(r >> 16);
}
__device__ __forceinline__ unsigned pack2bf(float a, float b) {
  return (unsigned)f2bfu(a) | ((unsigned)f2bfu(b) << 16);
}
__device__ __forceinline__ __bf16 bf_bits(unsigned short s) {
  union { unsigned short u; __bf16 b; } c; c.u = s; return c.b;
}

union FragBf { v16bf v; uint4 q[2]; };

// =========================================================================
// WMMA GEMM (all-bf16 data path):
//   D[M,N] = epilogue( alpha * A[M,K](bf16) @ B[N,K](bf16)^T )
// mode bit0: += R (f32 residual / prev_qk)
// mode bit1: squared-ReLU (before residual)
// mode bit2: bf16 output (else f32)
// Block: 128 threads (4 waves). Tile 128(M) x 64(N), K step 32.
// Each wave owns 32x64 -> 2x4 accumulators -> 8 v_wmma per K step.
// Staging: global_load_async_to_lds_b128 (ASYNCcnt) when available.
// Requires M%128==0, N%64==0, K%32==0 (true for all uses here).
// =========================================================================
__global__ __launch_bounds__(128) void k_gemm(
    const __bf16* __restrict__ A, const __bf16* __restrict__ Bm,
    const float* __restrict__ R, float* __restrict__ Df, __bf16* __restrict__ Db,
    int K, int lda, int ldb, int ldr, int ldd,
    float alpha, int mode)
{
  __shared__ alignas(16) __bf16 As[128][40];  // [m][k], 80B padded rows
  __shared__ alignas(16) __bf16 Bs[64][40];   // [n][k]

  const int tid    = threadIdx.x;
  const int lane   = tid & 31;
  const int wid    = tid >> 5;                // wave 0..3 -> rows wid*32
  const int m0     = blockIdx.y * 128;
  const int n0     = blockIdx.x * 64;
  const int laneHi = lane >> 4;
  const int lm     = lane & 15;

  const v8f vzero = {0.f,0.f,0.f,0.f,0.f,0.f,0.f,0.f};
  v8f acc[2][4] = {{vzero,vzero,vzero,vzero},{vzero,vzero,vzero,vzero}};

  const int br = tid >> 1;             // B row 0..63
  const int bh = (tid & 1) * 16;       // B k-offset 0/16

  for (int kk = 0; kk < K; kk += 32) {
    const __bf16* asrc = A  + (size_t)(m0 + tid) * lda + kk;
    const __bf16* bsrc = Bm + (size_t)(n0 + br) * ldb + kk + bh;
#if GEMM_ASYNC
    {
      AS1 v4i_vs* ga = (AS1 v4i_vs*)asrc;
      AS3 v4i_vs* la = (AS3 v4i_vs*)&As[tid][0];
      __builtin_amdgcn_global_load_async_to_lds_b128(ga, la, 0,  0);
      __builtin_amdgcn_global_load_async_to_lds_b128(ga, la, 16, 0);
      __builtin_amdgcn_global_load_async_to_lds_b128(ga, la, 32, 0);
      __builtin_amdgcn_global_load_async_to_lds_b128(ga, la, 48, 0);
      AS1 v4i_vs* gb = (AS1 v4i_vs*)bsrc;
      AS3 v4i_vs* lb = (AS3 v4i_vs*)&Bs[br][bh];
      __builtin_amdgcn_global_load_async_to_lds_b128(gb, lb, 0,  0);
      __builtin_amdgcn_global_load_async_to_lds_b128(gb, lb, 16, 0);
    }
#else
    {
      const uint4* src = (const uint4*)asrc;
      uint4* dst = (uint4*)&As[tid][0];
      dst[0] = src[0]; dst[1] = src[1]; dst[2] = src[2]; dst[3] = src[3];
      const uint4* sb = (const uint4*)bsrc;
      uint4* db = (uint4*)&Bs[br][bh];
      db[0] = sb[0]; db[1] = sb[1];
    }
#endif
    // unconditional speculative prefetch of the next K tile (harmless past end)
    __builtin_prefetch(asrc + 32, 0, 3);
    __builtin_prefetch(bsrc + 32, 0, 3);
#if GEMM_ASYNC
    WAIT_ASYNC();
#endif
    __syncthreads();

    // ---- fragments (contiguous k = 16*laneHi + s, same order for A and B) ----
    FragBf af[2], bfr[4];
    #pragma unroll
    for (int mi = 0; mi < 2; ++mi) {
      const uint4* p = (const uint4*)&As[wid*32 + mi*16 + lm][laneHi*16];
      af[mi].q[0] = p[0];
      af[mi].q[1] = p[1];              // +16B = +8 elements
    }
    #pragma unroll
    for (int ni = 0; ni < 4; ++ni) {
      const uint4* p = (const uint4*)&Bs[ni*16 + lm][laneHi*16];
      bfr[ni].q[0] = p[0];
      bfr[ni].q[1] = p[1];
    }
    #pragma unroll
    for (int mi = 0; mi < 2; ++mi)
      #pragma unroll
      for (int ni = 0; ni < 4; ++ni)
        acc[mi][ni] = __builtin_amdgcn_wmma_f32_16x16x32_bf16(
            false, af[mi].v, false, bfr[ni].v, (short)0, acc[mi][ni], false, false);
    __syncthreads();
  }

  // ---- epilogue ----
  #pragma unroll
  for (int mi = 0; mi < 2; ++mi) {
    #pragma unroll
    for (int ni = 0; ni < 4; ++ni) {
      const int gmb = m0 + wid*32 + mi*16 + laneHi*8;
      const int gn  = n0 + ni*16 + lm;
      #pragma unroll
      for (int r = 0; r < 8; ++r) {
        float v = acc[mi][ni][r] * alpha;
        if (mode & 2) { v = fmaxf(v, 0.f); v = v * v; }
        if (mode & 1) v += R[(size_t)(gmb + r) * ldr + gn];
        if (mode & 4) Db[(size_t)(gmb + r) * ldd + gn] = bf_bits(f2bfu(v));
        else          Df[(size_t)(gmb + r) * ldd + gn] = v;
      }
    }
  }
}

// =========================================================================
// Weight pre-pass: f32 [K,N] -> bf16 [N,K] (convert + transpose, LDS tiled)
// grid (N/32, K/32), block (32,8)
// =========================================================================
__global__ __launch_bounds__(256) void k_wT(
    const float* __restrict__ in, __bf16* __restrict__ out, int K, int N)
{
  __shared__ float tile[32][33];
  const int n0 = blockIdx.x * 32, k0 = blockIdx.y * 32;
  const int tx = threadIdx.x, ty = threadIdx.y;
  #pragma unroll
  for (int i = 0; i < 4; ++i)
    tile[ty + 8*i][tx] = in[(size_t)(k0 + ty + 8*i) * N + n0 + tx];
  __syncthreads();
  #pragma unroll
  for (int i = 0; i < 4; ++i)
    out[(size_t)(n0 + ty + 8*i) * K + k0 + tx] = bf_bits(f2bfu(tile[tx][ty + 8*i]));
}

// bf16 transpose per batch: v [B,T,F] -> vt [B,F,T].  grid (F/32, T/32, B)
__global__ __launch_bounds__(256) void k_tbf(
    const __bf16* __restrict__ in, __bf16* __restrict__ out)
{
  __shared__ __bf16 tile[32][34];
  const int b = blockIdx.z;
  const __bf16* ib = in  + (size_t)b * Tn * Fn;
  __bf16*       ob = out + (size_t)b * Fn * Tn;
  const int f0 = blockIdx.x * 32, t0 = blockIdx.y * 32;
  const int tx = threadIdx.x, ty = threadIdx.y;
  #pragma unroll
  for (int i = 0; i < 4; ++i)
    tile[ty + 8*i][tx] = ib[(size_t)(t0 + ty + 8*i) * Fn + f0 + tx];
  __syncthreads();
  #pragma unroll
  for (int i = 0; i < 4; ++i)
    ob[(size_t)(f0 + ty + 8*i) * Tn + t0 + tx] = tile[tx][ty + 8*i];
}

// =========================================================================
// Elementwise / norm kernels
// =========================================================================

// Instance-norm over (bins, C) per (b,t) + ReLU + channel projection.
// xc: [B, Cmax, F, T]  ->  x0: [B, T, F] (f32)
__global__ __launch_bounds__(256) void k_inorm_proj(
    const float* __restrict__ xc, int Cmax, int C,
    const float* __restrict__ inw, const float* __restrict__ inb,
    const float* __restrict__ projw, float* __restrict__ x0)
{
  __shared__ float sh[256];
  __shared__ float st[2];
  const int t = blockIdx.x, b = blockIdx.y, tid = threadIdx.x;
  const float* base = xc + (size_t)b * Cmax * FT + t;
  const int n = C * Fn;
  float s1 = 0.f, s2 = 0.f;
  for (int i = tid; i < n; i += 256) {
    const int c = i / Fn, f = i - c * Fn;
    const float v = base[(size_t)c * FT + (size_t)f * Tn];
    s1 += v; s2 += v * v;
  }
  sh[tid] = s1; __syncthreads();
  for (int s = 128; s > 0; s >>= 1) { if (tid < s) sh[tid] += sh[tid + s]; __syncthreads(); }
  if (tid == 0) st[0] = sh[0] / (float)n;
  __syncthreads();
  sh[tid] = s2; __syncthreads();
  for (int s = 128; s > 0; s >>= 1) { if (tid < s) sh[tid] += sh[tid + s]; __syncthreads(); }
  if (tid == 0) {
    const float m = st[0];
    st[1] = rsqrtf(fmaxf(sh[0] / (float)n - m * m, 0.f) + EPSf);
  }
  __syncthreads();
  const float mean = st[0], rs = st[1];
  const float ww = inw[t], bb = inb[t];
  for (int f = tid; f < Fn; f += 256) {
    float o = 0.f;
    for (int c = 0; c < C; ++c) {
      const float v = (base[(size_t)c * FT + (size_t)f * Tn] - mean) * rs * ww + bb;
      o += fmaxf(v, 0.f) * projw[c];
    }
    x0[((size_t)b * Tn + t) * Fn + f] = o;
  }
}

// Norm over F per (b,s), per-s scale/bias. u: [B,T,F] f32 -> h: [B,T,F] bf16
__global__ __launch_bounds__(256) void k_lnorm(
    const float* __restrict__ u, const float* __restrict__ wv,
    const float* __restrict__ bv, __bf16* __restrict__ h)
{
  __shared__ float sh[256];
  __shared__ float st[2];
  const int s = blockIdx.x, b = blockIdx.y, tid = threadIdx.x;
  const float* row = u + ((size_t)b * Tn + s) * Fn;
  float s1 = 0.f, s2 = 0.f;
  for (int f = tid; f < Fn; f += 256) { const float v = row[f]; s1 += v; s2 += v * v; }
  sh[tid] = s1; __syncthreads();
  for (int k = 128; k > 0; k >>= 1) { if (tid < k) sh[tid] += sh[tid + k]; __syncthreads(); }
  if (tid == 0) st[0] = sh[0] / (float)Fn;
  __syncthreads();
  sh[tid] = s2; __syncthreads();
  for (int k = 128; k > 0; k >>= 1) { if (tid < k) sh[tid] += sh[tid + k]; __syncthreads(); }
  if (tid == 0) {
    const float m = st[0];
    st[1] = rsqrtf(fmaxf(sh[0] / (float)Fn - m * m, 0.f) + EPSf);
  }
  __syncthreads();
  const float m = st[0], rs = st[1];
  const float ww = wv[s], bb = bv[s];
  unsigned* orow = (unsigned*)(h + ((size_t)b * Tn + s) * Fn);
  for (int f2 = tid; f2 < Fn / 2; f2 += 256) {
    const float va = (row[2*f2]   - m) * rs * ww + bb;
    const float vb = (row[2*f2+1] - m) * rs * ww + bb;
    orow[f2] = pack2bf(va, vb);
  }
}

// Row softmax, row length Tn=512, 256 threads (2 adjacent elems/thread).
// qk f32 -> a bf16
__global__ __launch_bounds__(256) void k_softmax(
    const float* __restrict__ qk, __bf16* __restrict__ a)
{
  __shared__ float sh[256];
  __shared__ float sv[2];
  const int row = blockIdx.x, tid = threadIdx.x;
  const float* src = qk + (size_t)row * Tn;
  const float v0 = src[2*tid], v1 = src[2*tid + 1];
  sh[tid] = fmaxf(v0, v1); __syncthreads();
  for (int s = 128; s > 0; s >>= 1) { if (tid < s) sh[tid] = fmaxf(sh[tid], sh[tid + s]); __syncthreads(); }
  if (tid == 0) sv[0] = sh[0];
  __syncthreads();
  const float m = sv[0];
  const float e0 = __expf(v0 - m), e1 = __expf(v1 - m);
  sh[tid] = e0 + e1; __syncthreads();
  for (int s = 128; s > 0; s >>= 1) { if (tid < s) sh[tid] += sh[tid + s]; __syncthreads(); }
  if (tid == 0) sv[1] = 1.f / sh[0];
  __syncthreads();
  const float inv = sv[1];
  unsigned* dst = (unsigned*)(a + (size_t)row * Tn);
  dst[tid] = pack2bf(e0 * inv, e1 * inv);
}

// Write x1 [B,T,F] (f32) into channel c of xc [B,Cmax,F,T] (transposed).
__global__ __launch_bounds__(256) void k_concat(
    const float* __restrict__ x1, float* __restrict__ xc, int Cmax, int c)
{
  const size_t idx = (size_t)blockIdx.x * 256 + threadIdx.x;
  if (idx >= (size_t)Bn * FT) return;
  const size_t b = idx / FT, r = idx % FT;
  const size_t f = r / Tn, t = r % Tn;
  xc[((size_t)b * Cmax + c) * FT + r] = x1[((size_t)b * Tn + t) * Fn + f];
}

// Copy input x [B,2,F,T] into channels 0..1 of xcE [B,4,F,T].
__global__ __launch_bounds__(256) void k_copy_in(
    const float* __restrict__ x, float* __restrict__ xc)
{
  const size_t idx = (size_t)blockIdx.x * 256 + threadIdx.x;
  if (idx >= (size_t)Bn * 2 * FT) return;
  const size_t b = idx / (2 * FT), r = idx % (2 * FT);
  const size_t c = r / FT, off = r % FT;
  xc[((size_t)b * 4 + c) * FT + off] = x[idx];
}

// BatchNorm batch stats per channel (grid = C blocks).
__global__ __launch_bounds__(256) void k_bnorm_stats(
    const float* __restrict__ xc, int Cmax,
    float* __restrict__ mout, float* __restrict__ rout)
{
  __shared__ float sh[256];
  __shared__ float st[1];
  const int c = blockIdx.x, tid = threadIdx.x;
  const size_t n = (size_t)Bn * FT;
  float s1 = 0.f, s2 = 0.f;
  for (size_t i = tid; i < n; i += 256) {
    const size_t b = i / FT, off = i % FT;
    const float v = xc[((size_t)b * Cmax + c) * FT + off];
    s1 += v; s2 += v * v;
  }
  sh[tid] = s1; __syncthreads();
  for (int s = 128; s > 0; s >>= 1) { if (tid < s) sh[tid] += sh[tid + s]; __syncthreads(); }
  if (tid == 0) { st[0] = sh[0] / (float)n; mout[c] = st[0]; }
  __syncthreads();
  sh[tid] = s2; __syncthreads();
  for (int s = 128; s > 0; s >>= 1) { if (tid < s) sh[tid] += sh[tid + s]; __syncthreads(); }
  if (tid == 0) {
    const float m = st[0];
    rout[c] = rsqrtf(fmaxf(sh[0] / (float)n - m * m, 0.f) + EPSf);
  }
}

__global__ void k_zero(float* __restrict__ p) {
  if (blockIdx.x == 0 && threadIdx.x == 0) *p = 0.f;
}

// Bottleneck: BN(4ch) + 4->1 projection + 1-D VQ nearest + STE write + loss.
__global__ __launch_bounds__(256) void k_bot_vq(
    const float* __restrict__ xc,
    const float* __restrict__ bm, const float* __restrict__ br,
    const float* __restrict__ bw, const float* __restrict__ bb,
    const float* __restrict__ pw, const float* __restrict__ pb,
    const float* __restrict__ code,
    float* __restrict__ xq, float* __restrict__ loss)
{
  __shared__ float cb[KCn];
  __shared__ float sh[256];
  const int tid = threadIdx.x;
  for (int i = tid; i < KCn; i += 256) cb[i] = code[i];
  __syncthreads();
  const size_t idx = (size_t)blockIdx.x * 256 + tid;
  float lcl = 0.f;
  if (idx < (size_t)Bn * FT) {
    const size_t b = idx / FT, off = idx % FT;
    float z = pb[0];
    #pragma unroll
    for (int c = 0; c < 4; ++c)
      z += ((xc[((size_t)b * 4 + c) * FT + off] - bm[c]) * br[c] * bw[c] + bb[c]) * pw[c];
    float bd = 3.4028235e38f, bz = 0.f;
    for (int k = 0; k < KCn; ++k) {
      float d = z - cb[k]; d *= d;
      if (d < bd) { bd = d; bz = cb[k]; }
    }
    xq[((size_t)b * 3 + 0) * FT + off] = bz;     // straight-through forward
    lcl = bd;
  }
  sh[tid] = lcl; __syncthreads();
  for (int s = 128; s > 0; s >>= 1) { if (tid < s) sh[tid] += sh[tid + s]; __syncthreads(); }
  if (tid == 0)
    atomicAdd(loss, sh[0] * ((1.0f + 0.25f) / (float)((size_t)Bn * FT)));
}

// Output: BN(3ch) + 3->2 projection + ReLU -> y [B,2,F,T]
__global__ __launch_bounds__(256) void k_out(
    const float* __restrict__ xq,
    const float* __restrict__ bm, const float* __restrict__ br,
    const float* __restrict__ bw, const float* __restrict__ bb,
    const float* __restrict__ ow, const float* __restrict__ obi,
    float* __restrict__ y)
{
  const size_t idx = (size_t)blockIdx.x * 256 + threadIdx.x;
  if (idx >= (size_t)Bn * FT) return;
  const size_t b = idx / FT, off = idx % FT;
  float a0 = obi[0], a1 = obi[1];
  #pragma unroll
  for (int c = 0; c < 3; ++c) {
    const float u = (xq[((size_t)b * 3 + c) * FT + off] - bm[c]) * br[c] * bw[c] + bb[c];
    a0 += u * ow[c * 2 + 0];
    a1 += u * ow[c * 2 + 1];
  }
  y[((size_t)b * 2 + 0) * FT + off] = fmaxf(a0, 0.f);
  y[((size_t)b * 2 + 1) * FT + off] = fmaxf(a1, 0.f);
}

// =========================================================================
// Host orchestration
// =========================================================================
struct WCache { __bf16 *wq, *wk, *wv, *wo, *f1, *f2; };

static inline void gemm(hipStream_t st, const __bf16* A, const __bf16* Bm,
                        const float* R, float* Df, __bf16* Db,
                        int M, int N, int K, int lda, int ldb, int ldr, int ldd,
                        float alpha, int mode)
{
  dim3 g(N / 64, M / 128), blk(128);
  k_gemm<<<g, blk, 0, st>>>(A, Bm, R, Df, Db, K, lda, ldb, ldr, ldd, alpha, mode);
}

// Per-block params (13 tensors): +0 in_norm_w +1 in_norm_b +2 in_proj_w
// +3 norm1_w +4 norm1_b +5 wq +6 wk +7 wv +8 wo +9 norm2_w +10 norm2_b +11 ff1 +12 ff2
static void run_block(hipStream_t st, void* const* d_in, int pb, const WCache& wc,
                      float* xc, int Cmax, int C,
                      const float* prevqk, float* qkout,
                      float* x0, float* x1,
                      __bf16* hbf, __bf16* qbf, __bf16* kbf, __bf16* vbf,
                      __bf16* vtbf, __bf16* obf, __bf16* abf, __bf16* fbf,
                      int outCh)
{
  auto P = [&](int i) { return (const float*)d_in[i]; };
  const dim3 b256(256);
  const int M = Bn * Tn;

  k_inorm_proj<<<dim3(Tn, Bn), b256, 0, st>>>(xc, Cmax, C, P(pb+0), P(pb+1), P(pb+2), x0);
  k_lnorm<<<dim3(Tn, Bn), b256, 0, st>>>(x0, P(pb+3), P(pb+4), hbf);

  gemm(st, hbf, wc.wq, nullptr, nullptr, qbf, M, Fn, Fn, Fn, Fn, 0, Fn, 1.f, 4);
  gemm(st, hbf, wc.wk, nullptr, nullptr, kbf, M, Fn, Fn, Fn, Fn, 0, Fn, 1.f, 4);
  gemm(st, hbf, wc.wv, nullptr, nullptr, vbf, M, Fn, Fn, Fn, Fn, 0, Fn, 1.f, 4);
  k_tbf<<<dim3(Fn/32, Tn/32, Bn), dim3(32, 8), 0, st>>>(vbf, vtbf);

  const float sc = 1.0f / 16.0f;     // 1/sqrt(256)
  for (int b = 0; b < Bn; ++b)
    for (int h = 0; h < HEADS; ++h) {
      const __bf16* qs = qbf + (size_t)b * Tn * Fn + h * DHn;
      const __bf16* ks = kbf + (size_t)b * Tn * Fn + h * DHn;
      float* dq = qkout + ((size_t)(b * HEADS + h) * Tn) * Tn;
      const float* rq = prevqk ? prevqk + ((size_t)(b * HEADS + h) * Tn) * Tn : nullptr;
      gemm(st, qs, ks, rq, dq, nullptr, Tn, Tn, DHn, Fn, Fn, Tn, Tn, sc, prevqk ? 1 : 0);
    }

  k_softmax<<<dim3(Bn * HEADS * Tn), b256, 0, st>>>(qkout, abf);

  for (int b = 0; b < Bn; ++b)
    for (int h = 0; h < HEADS; ++h) {
      const __bf16* as = abf + ((size_t)(b * HEADS + h) * Tn) * Tn;
      const __bf16* vs = vtbf + (size_t)b * Fn * Tn + (size_t)h * DHn * Tn;
      __bf16* os = obf + (size_t)b * Tn * Fn + h * DHn;
      gemm(st, as, vs, nullptr, nullptr, os, Tn, DHn, Tn, Tn, Tn, 0, Fn, 1.f, 4);
    }

  gemm(st, obf, wc.wo, x0, x1, nullptr, M, Fn, Fn, Fn, Fn, Fn, Fn, 1.f, 1);  // x1 = x0 + o@wo
  k_lnorm<<<dim3(Tn, Bn), b256, 0, st>>>(x1, P(pb+9), P(pb+10), hbf);
  gemm(st, hbf, wc.f1, nullptr, nullptr, fbf, M, FFDn, Fn, Fn, Fn, 0, FFDn, 1.f, 2 | 4);
  gemm(st, fbf, wc.f2, x1, x1, nullptr, M, Fn, FFDn, FFDn, FFDn, Fn, Fn, 1.f, 1);

  const size_t n = (size_t)Bn * FT;
  k_concat<<<dim3((unsigned)((n + 255) / 256)), b256, 0, st>>>(x1, xc, Cmax, outCh);
}

extern "C" void kernel_launch(void* const* d_in, const int* in_sizes, int n_in,
                              void* d_out, int out_size, void* d_ws, size_t ws_size,
                              hipStream_t stream)
{
  (void)in_sizes; (void)n_in; (void)out_size; (void)ws_size;
  auto P = [&](int i) { return (const float*)d_in[i]; };

  // Flat input index map (dict insertion order):
  // 0:x | enc blk b: 1+13b | dec blk b: 27+13b |
  // 53:bn_bot_w 54:bn_bot_b 55:bot_w 56:bot_b 57:codebook
  // 58:bn_out_w 59:bn_out_b 60:out_w 61:out_b

  // ---- workspace carve (256B aligned chunks) ----
  char* wp = (char*)d_ws;
  auto allocB = [&](size_t bytes) {
    char* p = wp; wp += (bytes + 255) & ~(size_t)255; return (void*)p;
  };
  float* xcE  = (float*)allocB((size_t)Bn * 4 * FT * 4);
  float* xcD  = (float*)allocB((size_t)Bn * 3 * FT * 4);
  float* x0   = (float*)allocB((size_t)Bn * Tn * Fn * 4);
  float* x1   = (float*)allocB((size_t)Bn * Tn * Fn * 4);
  float* qkA  = (float*)allocB((size_t)Bn * HEADS * Tn * Tn * 4);
  float* qkB  = (float*)allocB((size_t)Bn * HEADS * Tn * Tn * 4);
  float* statm = (float*)allocB(8 * 4);
  float* statr = (float*)allocB(8 * 4);
  __bf16* hbf  = (__bf16*)allocB((size_t)Bn * Tn * Fn * 2);
  __bf16* qbf  = (__bf16*)allocB((size_t)Bn * Tn * Fn * 2);
  __bf16* kbf  = (__bf16*)allocB((size_t)Bn * Tn * Fn * 2);
  __bf16* vbf  = (__bf16*)allocB((size_t)Bn * Tn * Fn * 2);
  __bf16* vtbf = (__bf16*)allocB((size_t)Bn * Fn * Tn * 2);
  __bf16* obf  = (__bf16*)allocB((size_t)Bn * Tn * Fn * 2);
  __bf16* abf  = (__bf16*)allocB((size_t)Bn * HEADS * Tn * Tn * 2);
  __bf16* fbf  = (__bf16*)allocB((size_t)Bn * Tn * FFDn * 2);

  // bf16 weight cache: convert + transpose all block weights once
  WCache wc[4];
  const int pbase[4] = { 1, 14, 27, 40 };
  for (int blk = 0; blk < 4; ++blk) {
    wc[blk].wq = (__bf16*)allocB((size_t)Fn * Fn * 2);
    wc[blk].wk = (__bf16*)allocB((size_t)Fn * Fn * 2);
    wc[blk].wv = (__bf16*)allocB((size_t)Fn * Fn * 2);
    wc[blk].wo = (__bf16*)allocB((size_t)Fn * Fn * 2);
    wc[blk].f1 = (__bf16*)allocB((size_t)Fn * FFDn * 2);
    wc[blk].f2 = (__bf16*)allocB((size_t)FFDn * Fn * 2);
    const int pb = pbase[blk];
    const dim3 tb(32, 8);
    k_wT<<<dim3(Fn/32,   Fn/32),   tb, 0, stream>>>(P(pb+5),  wc[blk].wq, Fn,   Fn);
    k_wT<<<dim3(Fn/32,   Fn/32),   tb, 0, stream>>>(P(pb+6),  wc[blk].wk, Fn,   Fn);
    k_wT<<<dim3(Fn/32,   Fn/32),   tb, 0, stream>>>(P(pb+7),  wc[blk].wv, Fn,   Fn);
    k_wT<<<dim3(Fn/32,   Fn/32),   tb, 0, stream>>>(P(pb+8),  wc[blk].wo, Fn,   Fn);
    k_wT<<<dim3(FFDn/32, Fn/32),   tb, 0, stream>>>(P(pb+11), wc[blk].f1, Fn,   FFDn); // [F,FFD]->[FFD,F]
    k_wT<<<dim3(Fn/32,   FFDn/32), tb, 0, stream>>>(P(pb+12), wc[blk].f2, FFDn, Fn);   // [FFD,F]->[F,FFD]
  }

  const dim3 b256(256);
  const size_t nBFT = (size_t)Bn * FT;

  // ---- encoder ----
  {
    const size_t n = (size_t)Bn * 2 * FT;
    k_copy_in<<<dim3((unsigned)((n + 255) / 256)), b256, 0, stream>>>(P(0), xcE);
  }
  float* qkbuf[2] = { qkA, qkB };
  const float* prev = nullptr;
  for (int blk = 0; blk < 2; ++blk) {
    run_block(stream, d_in, pbase[blk], wc[blk], xcE, 4, 2 + blk, prev, qkbuf[blk],
              x0, x1, hbf, qbf, kbf, vbf, vtbf, obf, abf, fbf, 2 + blk);
    prev = qkbuf[blk];
  }

  // ---- bottleneck + VQ ----
  k_bnorm_stats<<<dim3(4), b256, 0, stream>>>(xcE, 4, statm, statr);
  float* dloss = (float*)d_out + (size_t)Bn * 2 * FT;
  k_zero<<<1, 1, 0, stream>>>(dloss);
  k_bot_vq<<<dim3((unsigned)((nBFT + 255) / 256)), b256, 0, stream>>>(
      xcE, statm, statr, P(53), P(54), P(55), P(56), P(57), xcD, dloss);

  // ---- decoder ----
  prev = nullptr;
  for (int blk = 0; blk < 2; ++blk) {
    run_block(stream, d_in, pbase[2 + blk], wc[2 + blk], xcD, 3, 1 + blk, prev, qkbuf[blk],
              x0, x1, hbf, qbf, kbf, vbf, vtbf, obf, abf, fbf, 1 + blk);
    prev = qkbuf[blk];
  }

  // ---- output head ----
  k_bnorm_stats<<<dim3(3), b256, 0, stream>>>(xcD, 3, statm, statr);
  k_out<<<dim3((unsigned)((nBFT + 255) / 256)), b256, 0, stream>>>(
      xcD, statm, statr, P(58), P(59), P(60), P(61), (float*)d_out);
}
